// FrameTransformer_29489245454556
// MI455X (gfx1250) — compile-verified, matrix operand order checked by reading, not run on hardware
//
#include <hip/hip_runtime.h>
#include <hip/hip_bf16.h>

// ---------------------------------------------------------------------------
// CDNA5 (gfx1250) implementation: every matmul runs on v_wmma_f32_16x16x32_bf16.
// ---------------------------------------------------------------------------

typedef __bf16 bf16_t;
typedef __attribute__((ext_vector_type(16))) __bf16 v16bf;
typedef __attribute__((ext_vector_type(8)))  __bf16 v8bf;
typedef __attribute__((ext_vector_type(8)))  float  v8f;

#define BB 4
#define TT 1024
#define FFULL 1025
#define BINS 1024
#define HEADS 4
#define DHEAD 256
#define FFN 2048

enum { EP_BIAS = 1, EP_RELU = 2, EP_RELUSQ = 4 };

// Load one WMMA bf16 fragment half-pair: lane holds two contiguous 8xbf16 chunks.
__device__ __forceinline__ v16bf load_frag(const bf16_t* __restrict__ p) {
  union { v16bf v; v8bf h[2]; } u;
  u.h[0] = *(const v8bf*)(p);
  u.h[1] = *(const v8bf*)(p + 16);
  return u.v;
}

// Generic batched GEMM: C[z][m][n] = sum_k A[z][m][k] * Bw[z][n][k]  (+epilogue)
// z offsets: off = (z/zdiv)*S0 + (z%zdiv)*S1 for each operand.
// Wave tile 32x64 (2x4 of 16x16), block = 4 waves -> 128x64.
__global__ __launch_bounds__(128) void wmma_gemm(
    const bf16_t* __restrict__ A, const bf16_t* __restrict__ Bw,
    float* __restrict__ C, bf16_t* __restrict__ C2,
    const float* __restrict__ bias,
    int M, int N, int K, int lda, int ldb, int ldc,
    long aS0, long aS1, long bS0, long bS1, long cS0, long cS1,
    int zdiv, int flags)
{
  const int z  = blockIdx.z;
  const int zh = z / zdiv, zl = z % zdiv;
  A  += (size_t)zh * aS0 + (size_t)zl * aS1;
  Bw += (size_t)zh * bS0 + (size_t)zl * bS1;
  const size_t coff = (size_t)zh * cS0 + (size_t)zl * cS1;

  const int lane = threadIdx.x & 31;
  const int wave = threadIdx.x >> 5;
  const int m0   = blockIdx.y * 128 + wave * 32;
  const int n0   = blockIdx.x * 64;
  const int half = lane >> 4;
  const int l16  = lane & 15;

  v8f acc[2][4];
#pragma unroll
  for (int i = 0; i < 2; ++i)
#pragma unroll
    for (int j = 0; j < 4; ++j)
#pragma unroll
      for (int d = 0; d < 8; ++d) acc[i][j][d] = 0.0f;

  const bf16_t* arow0 = A + (size_t)(m0 + l16) * lda + (half << 3);
  const bf16_t* arow1 = arow0 + (size_t)16 * lda;
  const bf16_t* brow0 = Bw + (size_t)(n0 +  0 + l16) * ldb + (half << 3);
  const bf16_t* brow1 = Bw + (size_t)(n0 + 16 + l16) * ldb + (half << 3);
  const bf16_t* brow2 = Bw + (size_t)(n0 + 32 + l16) * ldb + (half << 3);
  const bf16_t* brow3 = Bw + (size_t)(n0 + 48 + l16) * ldb + (half << 3);

  for (int k = 0; k < K; k += 32) {
    v16bf a0 = load_frag(arow0 + k);
    v16bf a1 = load_frag(arow1 + k);
    v16bf b0 = load_frag(brow0 + k);
    v16bf b1 = load_frag(brow1 + k);
    v16bf b2 = load_frag(brow2 + k);
    v16bf b3 = load_frag(brow3 + k);
    acc[0][0] = __builtin_amdgcn_wmma_f32_16x16x32_bf16(false, a0, false, b0, (short)0, acc[0][0], false, false);
    acc[0][1] = __builtin_amdgcn_wmma_f32_16x16x32_bf16(false, a0, false, b1, (short)0, acc[0][1], false, false);
    acc[0][2] = __builtin_amdgcn_wmma_f32_16x16x32_bf16(false, a0, false, b2, (short)0, acc[0][2], false, false);
    acc[0][3] = __builtin_amdgcn_wmma_f32_16x16x32_bf16(false, a0, false, b3, (short)0, acc[0][3], false, false);
    acc[1][0] = __builtin_amdgcn_wmma_f32_16x16x32_bf16(false, a1, false, b0, (short)0, acc[1][0], false, false);
    acc[1][1] = __builtin_amdgcn_wmma_f32_16x16x32_bf16(false, a1, false, b1, (short)0, acc[1][1], false, false);
    acc[1][2] = __builtin_amdgcn_wmma_f32_16x16x32_bf16(false, a1, false, b2, (short)0, acc[1][2], false, false);
    acc[1][3] = __builtin_amdgcn_wmma_f32_16x16x32_bf16(false, a1, false, b3, (short)0, acc[1][3], false, false);
  }

#pragma unroll
  for (int j = 0; j < 4; ++j) {
    const int n = n0 + j * 16 + l16;
    const float bv = (flags & EP_BIAS) ? bias[n] : 0.0f;
#pragma unroll
    for (int i = 0; i < 2; ++i) {
      const int mbase = m0 + i * 16 + (half << 3);
#pragma unroll
      for (int d = 0; d < 8; ++d) {
        float v = acc[i][j][d] + bv;
        if (flags & EP_RELU)   v = fmaxf(v, 0.0f);
        if (flags & EP_RELUSQ) { v = fmaxf(v, 0.0f); v = v * v; }
        const size_t idx = coff + (size_t)(mbase + d) * ldc + n;
        if (C)  C[idx]  = v;
        if (C2) C2[idx] = (bf16_t)v;
      }
    }
  }
}

// fp32 -> bf16 weight conversion
__global__ __launch_bounds__(256) void convert_kernel(const float* __restrict__ s,
                                                      bf16_t* __restrict__ d, int n) {
  int i = blockIdx.x * 256 + threadIdx.x;
  if (i < n) d[i] = (bf16_t)s[i];
}

// erT[r][d] = er[d][r]  (256x1024 -> 1024x256, bf16)
__global__ __launch_bounds__(256) void ert_kernel(const float* __restrict__ er,
                                                  bf16_t* __restrict__ erT) {
  int i = blockIdx.x * 256 + threadIdx.x;   // i = r*256 + d
  int r = i >> 8, d = i & 255;
  erT[i] = (bf16_t)er[(size_t)d * 1024 + r];
}

// x[b,t,f] = w0*src[b,0,f,t] + w1*src[b,1,f,t]  (tiled transpose)
__global__ __launch_bounds__(256) void input_kernel(const float* __restrict__ src,
                                                    const float* __restrict__ inw,
                                                    float* __restrict__ x) {
  __shared__ float ts[32][33];
  const int b = blockIdx.z;
  const int f0 = blockIdx.x * 32, t0 = blockIdx.y * 32;
  const float w0 = inw[0], w1 = inw[1];
  const size_t sb = (size_t)b * 2 * FFULL * TT;
  const int tc = threadIdx.x & 31, tr = threadIdx.x >> 5;
#pragma unroll
  for (int q = 0; q < 4; ++q) {
    int fl = tr + q * 8;
    size_t o = sb + (size_t)(f0 + fl) * TT + t0 + tc;
    ts[fl][tc] = w0 * src[o] + w1 * src[o + (size_t)FFULL * TT];
  }
  __syncthreads();
#pragma unroll
  for (int q = 0; q < 4; ++q) {
    int tl = tr + q * 8;
    x[((size_t)b * TT + t0 + tl) * BINS + f0 + tc] = ts[tc][tl];
  }
}

// LayerNorm over W=1024, one row per block. Optional residual add (written back).
__global__ __launch_bounds__(256) void ln_kernel(
    float* __restrict__ x, const float* __restrict__ add, int writeback,
    const float* __restrict__ g, const float* __restrict__ b,
    float* __restrict__ outf, bf16_t* __restrict__ outbf)
{
  const size_t row = blockIdx.x;
  float* xr = x + row * 1024;
  const float* ar = add ? add + row * 1024 : nullptr;
  __shared__ float r1[256], r2[256];
  float vloc[4], s = 0.f, s2 = 0.f;
#pragma unroll
  for (int q = 0; q < 4; ++q) {
    int c = threadIdx.x + q * 256;
    float v = xr[c];
    if (ar) v += ar[c];
    vloc[q] = v; s += v; s2 += v * v;
  }
  r1[threadIdx.x] = s; r2[threadIdx.x] = s2;
  __syncthreads();
  for (int o = 128; o > 0; o >>= 1) {
    if (threadIdx.x < o) { r1[threadIdx.x] += r1[threadIdx.x + o]; r2[threadIdx.x] += r2[threadIdx.x + o]; }
    __syncthreads();
  }
  const float mean = r1[0] * (1.0f / 1024.0f);
  const float var  = r2[0] * (1.0f / 1024.0f) - mean * mean;
  const float inv  = rsqrtf(var + 1e-5f);
#pragma unroll
  for (int q = 0; q < 4; ++q) {
    int c = threadIdx.x + q * 256;
    float v = vloc[q];
    if (writeback) xr[c] = v;
    float o = (v - mean) * inv * g[c] + b[c];
    if (outf)  outf[row * 1024 + c] = o;
    if (outbf) outbf[row * 1024 + c] = (bf16_t)o;
  }
}

// LN3: row (b,i): val[j] = hL[b,i,j] + (j<512 ? hR0[b,j,i] : 0); LN -> h3 fp32
__global__ __launch_bounds__(256) void ln3_kernel(
    const float* __restrict__ hL, const float* __restrict__ hR0,
    const float* __restrict__ g, const float* __restrict__ b,
    float* __restrict__ h3)
{
  const int row = blockIdx.x;
  const int bb = row >> 10, i = row & 1023;
  __shared__ float r1[256], r2[256];
  float vloc[4], s = 0.f, s2 = 0.f;
#pragma unroll
  for (int q = 0; q < 4; ++q) {
    int c = threadIdx.x + q * 256;
    float v = hL[((size_t)bb * 1024 + i) * 1024 + c];
    if (c < 512) v += hR0[((size_t)bb * 512 + c) * 1024 + i];
    vloc[q] = v; s += v; s2 += v * v;
  }
  r1[threadIdx.x] = s; r2[threadIdx.x] = s2;
  __syncthreads();
  for (int o = 128; o > 0; o >>= 1) {
    if (threadIdx.x < o) { r1[threadIdx.x] += r1[threadIdx.x + o]; r2[threadIdx.x] += r2[threadIdx.x + o]; }
    __syncthreads();
  }
  const float mean = r1[0] * (1.0f / 1024.0f);
  const float var  = r2[0] * (1.0f / 1024.0f) - mean * mean;
  const float inv  = rsqrtf(var + 1e-5f);
#pragma unroll
  for (int q = 0; q < 4; ++q) {
    int c = threadIdx.x + q * 256;
    h3[(size_t)row * 1024 + c] = (vloc[q] - mean) * inv * g[c] + b[c];
  }
}

// GLU gate: x += g2[:, :1024] * sigmoid(g2[:, 1024:])
__global__ __launch_bounds__(256) void glu_kernel(float* __restrict__ x,
                                                  const float* __restrict__ g2) {
  size_t idx = (size_t)blockIdx.x * 256 + threadIdx.x;  // B*T*1024 total
  size_t r = idx >> 10, f = idx & 1023;
  float a  = g2[r * 2048 + f];
  float gt = g2[r * 2048 + 1024 + f];
  x[idx] += a * (1.0f / (1.0f + __expf(-gt)));
}

// Depthwise 11-tap conv over frequency (channel = t), output transposed (f,t), bf16.
__global__ __launch_bounds__(256) void dconvL_kernel(const float* __restrict__ h,
                                                     const float* __restrict__ w,
                                                     bf16_t* __restrict__ XcT) {
  __shared__ float hs[32][43];
  const int b = blockIdx.z;
  const int f0 = blockIdx.x * 32, t0 = blockIdx.y * 32;
  for (int idx = threadIdx.x; idx < 32 * 42; idx += 256) {
    int r = idx / 42, c = idx % 42;
    int gf = f0 + c - 5;
    hs[r][c] = (gf >= 0 && gf < 1024) ? h[((size_t)b * TT + t0 + r) * BINS + gf] : 0.0f;
  }
  __syncthreads();
  const int tl = threadIdx.x & 31, fr = threadIdx.x >> 5;
  float wr[11];
#pragma unroll
  for (int j = 0; j < 11; ++j) wr[j] = w[(t0 + tl) * 11 + j];
#pragma unroll
  for (int q = 0; q < 4; ++q) {
    int fl = fr + q * 8;
    float s = 0.f;
#pragma unroll
    for (int j = 0; j < 11; ++j) s += hs[tl][fl + j] * wr[j];
    XcT[((size_t)b * BINS + f0 + fl) * TT + t0 + tl] = (bf16_t)s;
  }
}

// Depthwise 7-tap conv over time (channel = f, pad 3), optional bias, bf16 out (t,f).
__global__ __launch_bounds__(256) void dconv7_kernel(const float* __restrict__ in,
                                                     const float* __restrict__ w,
                                                     const float* __restrict__ bias,
                                                     bf16_t* __restrict__ out) {
  const int row = blockIdx.x;          // b*1024 + t
  const int b = row >> 10, t = row & 1023;
#pragma unroll
  for (int q = 0; q < 4; ++q) {
    int f = threadIdx.x + q * 256;
    float s = bias ? bias[f] : 0.0f;
#pragma unroll
    for (int j = 0; j < 7; ++j) {
      int tt = t + j - 3;
      if (tt >= 0 && tt < 1024)
        s += in[((size_t)b * 1024 + tt) * 1024 + f] * w[f * 7 + j];
    }
    out[(size_t)row * 1024 + f] = (bf16_t)s;
  }
}

// vT[bh][d][t] = zv[b][t][h*256+d]
__global__ __launch_bounds__(256) void vtrans_kernel(const bf16_t* __restrict__ zv,
                                                     bf16_t* __restrict__ vT) {
  __shared__ bf16_t ts[32][33];
  const int bh = blockIdx.z;
  const int b = bh >> 2, hh = bh & 3;
  const int d0 = blockIdx.x * 32, t0 = blockIdx.y * 32;
  const int tc = threadIdx.x & 31, tr = threadIdx.x >> 5;
#pragma unroll
  for (int q = 0; q < 4; ++q) {
    int tl = tr + q * 8;
    ts[tl][tc] = zv[((size_t)b * TT + t0 + tl) * BINS + hh * DHEAD + d0 + tc];
  }
  __syncthreads();
#pragma unroll
  for (int q = 0; q < 4; ++q) {
    int dl = tr + q * 8;
    vT[((size_t)bh * DHEAD + d0 + dl) * TT + t0 + tc] = ts[tc][dl];
  }
}

// softmax over j: S[j] = (qk[row,j] + p[bh][j][i]) / 32  -> bf16 attn weights
__global__ __launch_bounds__(256) void softmax_kernel(const float* __restrict__ qk,
                                                      const float* __restrict__ p,
                                                      bf16_t* __restrict__ out) {
  const int row = blockIdx.x;          // (b*4+h)*1024 + i
  const int bh = row >> 10, i = row & 1023;
  const float* qr = qk + (size_t)row * 1024;
  const float* pr = p + (size_t)bh * 1024 * 1024 + i;
  __shared__ float red[256];
  float v[4], mx = -1e30f;
#pragma unroll
  for (int q = 0; q < 4; ++q) {
    int j = threadIdx.x + q * 256;
    v[q] = (qr[j] + pr[(size_t)j * 1024]) * 0.03125f;
    mx = fmaxf(mx, v[q]);
  }
  red[threadIdx.x] = mx;
  __syncthreads();
  for (int o = 128; o > 0; o >>= 1) {
    if (threadIdx.x < o) red[threadIdx.x] = fmaxf(red[threadIdx.x], red[threadIdx.x + o]);
    __syncthreads();
  }
  mx = red[0];
  __syncthreads();
  float s = 0.f;
#pragma unroll
  for (int q = 0; q < 4; ++q) { v[q] = __expf(v[q] - mx); s += v[q]; }
  red[threadIdx.x] = s;
  __syncthreads();
  for (int o = 128; o > 0; o >>= 1) {
    if (threadIdx.x < o) red[threadIdx.x] += red[threadIdx.x + o];
    __syncthreads();
  }
  const float inv = 1.0f / red[0];
#pragma unroll
  for (int q = 0; q < 4; ++q) {
    int j = threadIdx.x + q * 256;
    out[(size_t)row * 1024 + j] = (bf16_t)(v[q] * inv);
  }
}

// y[b,o,f,t] = clip(ow[o,0]*s0 + ow[o,1]*s1 + ow[o,2]*(x+ff)[b,t,f], 0, 6)/6
__global__ __launch_bounds__(256) void final_kernel(const float* __restrict__ src,
                                                    const float* __restrict__ x,
                                                    const float* __restrict__ ff,
                                                    const float* __restrict__ ow,
                                                    float* __restrict__ out) {
  __shared__ float ts[32][33];
  const int b = blockIdx.z;
  const int f0 = blockIdx.x * 32, t0 = blockIdx.y * 32;
  const int tc = threadIdx.x & 31, tr = threadIdx.x >> 5;
#pragma unroll
  for (int q = 0; q < 4; ++q) {
    int tl = tr + q * 8;
    size_t o = ((size_t)b * TT + t0 + tl) * BINS + f0 + tc;
    ts[tl][tc] = x[o] + ff[o];
  }
  __syncthreads();
  const size_t sb = (size_t)b * 2 * FFULL * TT;
#pragma unroll
  for (int q = 0; q < 4; ++q) {
    int fl = tr + q * 8;
    int f = f0 + fl, t = t0 + tc;
    float s0 = src[sb + (size_t)f * TT + t];
    float s1 = src[sb + (size_t)FFULL * TT + (size_t)f * TT + t];
    float xv = ts[tc][fl];
#pragma unroll
    for (int o = 0; o < 2; ++o) {
      float y = ow[o * 3 + 0] * s0 + ow[o * 3 + 1] * s1 + ow[o * 3 + 2] * xv;
      y = fminf(fmaxf(y, 0.0f), 6.0f) * (1.0f / 6.0f);
      out[sb + (size_t)o * FFULL * TT + (size_t)f * TT + t] = y;
    }
  }
}

// duplicate frequency row 1023 into row 1024
__global__ __launch_bounds__(256) void lastrow_kernel(float* __restrict__ out) {
  int idx = blockIdx.x * 256 + threadIdx.x;   // B*2*1024
  int bo = idx >> 10, t = idx & 1023;
  out[(size_t)bo * FFULL * TT + (size_t)1024 * TT + t] =
      out[(size_t)bo * FFULL * TT + (size_t)1023 * TT + t];
}

extern "C" void kernel_launch(void* const* d_in, const int* in_sizes, int n_in,
                              void* d_out, int out_size, void* d_ws, size_t ws_size,
                              hipStream_t stream) {
  (void)in_sizes; (void)n_in; (void)out_size; (void)ws_size;
  const float* src    = (const float*)d_in[0];
  const float* in_w   = (const float*)d_in[1];
  const float* ln1_g  = (const float*)d_in[2];
  const float* ln1_b  = (const float*)d_in[3];
  const float* glu_w  = (const float*)d_in[4];
  const float* ln2_g  = (const float*)d_in[5];
  const float* ln2_b  = (const float*)d_in[6];
  const float* c1L_dw = (const float*)d_in[7];
  const float* c1L_pw = (const float*)d_in[8];
  const float* c1R_dw = (const float*)d_in[9];
  const float* c1R_pw = (const float*)d_in[10];
  const float* ln3_g  = (const float*)d_in[11];
  const float* ln3_b  = (const float*)d_in[12];
  const float* c1M_dw = (const float*)d_in[13];
  const float* c1M_pw = (const float*)d_in[14];
  const float* ln4_g  = (const float*)d_in[15];
  const float* ln4_b  = (const float*)d_in[16];
  const float* q_w    = (const float*)d_in[17];
  const float* q_b    = (const float*)d_in[18];
  const float* qc_w   = (const float*)d_in[19];
  const float* qc_b   = (const float*)d_in[20];
  const float* k_w    = (const float*)d_in[21];
  const float* k_b    = (const float*)d_in[22];
  const float* v_w    = (const float*)d_in[23];
  const float* v_b    = (const float*)d_in[24];
  const float* o_w    = (const float*)d_in[25];
  const float* o_b    = (const float*)d_in[26];
  const float* er     = (const float*)d_in[27];
  const float* ln5_g  = (const float*)d_in[28];
  const float* ln5_b  = (const float*)d_in[29];
  const float* c2_w   = (const float*)d_in[30];
  const float* c3_w   = (const float*)d_in[31];
  const float* out_w  = (const float*)d_in[32];
  float* out = (float*)d_out;

  char* ws = (char*)d_ws;
  size_t off = 0;
  auto af32 = [&](size_t n) { float* p = (float*)(ws + off); off += ((n * 4 + 255) & ~(size_t)255); return p; };
  auto abf  = [&](size_t n) { bf16_t* p = (bf16_t*)(ws + off); off += ((n * 2 + 255) & ~(size_t)255); return p; };

  float* x    = af32(4194304);   // (B*T, 1024) residual stream
  float* h    = af32(4194304);   // ln2 out fp32; reused as y (qkv pre-conv)
  float* g2   = af32(8388608);   // GLU out; reused: h4 = g2[0:4M], ff = g2[4M:8M]
  float* hL   = af32(4194304);
  float* hR0  = af32(2097152);
  float* h3   = af32(4194304);   // reused as o_out
  float* qk   = af32(16777216);  // (B,4,T,T)
  float* pr   = af32(16777216);  // (B,4,T,1024) relative positions
  bf16_t* wb_glu = abf(2097152);
  bf16_t* wb_c1L = abf(1048576);
  bf16_t* wb_c1R = abf(524288);
  bf16_t* wb_c1M = abf(1048576);
  bf16_t* wb_q   = abf(1048576);
  bf16_t* wb_k   = abf(1048576);
  bf16_t* wb_v   = abf(1048576);
  bf16_t* wb_o   = abf(1048576);
  bf16_t* wb_c2  = abf(2097152);
  bf16_t* wb_c3  = abf(2097152);
  bf16_t* wb_erT = abf(262144);
  bf16_t* h_bf   = abf(4194304); // ln1/ln4/ln5 bf16 (sequential reuse)
  bf16_t* xp_bf  = abf(4194304); // XcT / XrT / XmT / vT (sequential reuse)
  bf16_t* zq_bf  = abf(4194304); // reused as o_concat bf16
  bf16_t* zk_bf  = abf(4194304);
  bf16_t* zv_bf  = abf(4194304);
  bf16_t* A_bf   = abf(16777216); // attention weights; reused as FFN-mid bf16
  float* h4 = g2;
  float* ff = g2 + 4194304;
  float* y  = h;
  float* o_out = h3;
  bf16_t* vT_bf = xp_bf;
  bf16_t* oc_bf = zq_bf;
  bf16_t* m_bf  = A_bf;

  auto cvt = [&](const float* s, bf16_t* d, int n) {
    convert_kernel<<<dim3((n + 255) / 256), dim3(256), 0, stream>>>(s, d, n);
  };
  auto gemm = [&](const bf16_t* A, const bf16_t* Bw, float* C, bf16_t* C2,
                  const float* bias, int M, int N, int K, int lda, int ldb, int ldc,
                  long aS0, long aS1, long bS0, long bS1, long cS0, long cS1,
                  int zdiv, int Z, int flags) {
    wmma_gemm<<<dim3(N / 64, M / 128, Z), dim3(128), 0, stream>>>(
        A, Bw, C, C2, bias, M, N, K, lda, ldb, ldc,
        aS0, aS1, bS0, bS1, cS0, cS1, zdiv, flags);
  };

  // weight conversions (deterministic, cheap)
  cvt(glu_w,  wb_glu, 2097152);
  cvt(c1L_pw, wb_c1L, 1048576);
  cvt(c1R_pw, wb_c1R, 524288);
  cvt(c1M_pw, wb_c1M, 1048576);
  cvt(q_w,    wb_q,   1048576);
  cvt(k_w,    wb_k,   1048576);
  cvt(v_w,    wb_v,   1048576);
  cvt(o_w,    wb_o,   1048576);
  cvt(c2_w,   wb_c2,  2097152);
  cvt(c3_w,   wb_c3,  2097152);
  ert_kernel<<<dim3(1024), dim3(256), 0, stream>>>(er, wb_erT);

  // stage 0: input channel-mix + transpose
  input_kernel<<<dim3(32, 32, BB), dim3(256), 0, stream>>>(src, in_w, x);

  // ln1 -> GLU GEMM -> gate
  ln_kernel<<<dim3(4096), dim3(256), 0, stream>>>(x, nullptr, 0, ln1_g, ln1_b, nullptr, h_bf);
  gemm(h_bf, wb_glu, g2, nullptr, nullptr, 4096, 2048, 1024, 1024, 1024, 2048,
       0, 0, 0, 0, 0, 0, 1, 1, 0);
  glu_kernel<<<dim3(16384), dim3(256), 0, stream>>>(x, g2);

  // ln2 -> conv branches
  ln_kernel<<<dim3(4096), dim3(256), 0, stream>>>(x, nullptr, 0, ln2_g, ln2_b, h, nullptr);
  dconvL_kernel<<<dim3(32, 32, BB), dim3(256), 0, stream>>>(h, c1L_dw, xp_bf);
  gemm(wb_c1L, xp_bf, hL, nullptr, nullptr, 1024, 1024, 1024, 1024, 1024, 1024,
       0, 0, 1048576, 0, 1048576, 0, 1, BB, EP_RELU);
  dconv7_kernel<<<dim3(4096), dim3(256), 0, stream>>>(h, c1R_dw, nullptr, xp_bf);
  gemm(wb_c1R, xp_bf, hR0, nullptr, nullptr, 512, 1024, 1024, 1024, 1024, 1024,
       0, 0, 1048576, 0, 524288, 0, 1, BB, EP_RELU);
  ln3_kernel<<<dim3(4096), dim3(256), 0, stream>>>(hL, hR0, ln3_g, ln3_b, h3);
  dconv7_kernel<<<dim3(4096), dim3(256), 0, stream>>>(h3, c1M_dw, nullptr, xp_bf);
  gemm(xp_bf, wb_c1M, h4, nullptr, nullptr, 4096, 1024, 1024, 1024, 1024, 1024,
       0, 0, 0, 0, 0, 0, 1, 1, 0);

  // ln4 (x += h4 residual) -> attention
  ln_kernel<<<dim3(4096), dim3(256), 0, stream>>>(x, h4, 1, ln4_g, ln4_b, nullptr, h_bf);
  gemm(h_bf, wb_q, y, nullptr, q_b, 4096, 1024, 1024, 1024, 1024, 1024,
       0, 0, 0, 0, 0, 0, 1, 1, EP_BIAS);
  dconv7_kernel<<<dim3(4096), dim3(256), 0, stream>>>(y, qc_w, qc_b, zq_bf);
  gemm(h_bf, wb_k, y, nullptr, k_b, 4096, 1024, 1024, 1024, 1024, 1024,
       0, 0, 0, 0, 0, 0, 1, 1, EP_BIAS);
  dconv7_kernel<<<dim3(4096), dim3(256), 0, stream>>>(y, qc_w, qc_b, zk_bf);
  gemm(h_bf, wb_v, y, nullptr, v_b, 4096, 1024, 1024, 1024, 1024, 1024,
       0, 0, 0, 0, 0, 0, 1, 1, EP_BIAS);
  dconv7_kernel<<<dim3(4096), dim3(256), 0, stream>>>(y, qc_w, qc_b, zv_bf);
  vtrans_kernel<<<dim3(8, 32, 16), dim3(256), 0, stream>>>(zv_bf, vT_bf);

  // q@k^T and q@er (16 batched heads)
  gemm(zq_bf, zk_bf, qk, nullptr, nullptr, 1024, 1024, 256, 1024, 1024, 1024,
       1048576, 256, 1048576, 256, 4194304, 1048576, 4, 16, 0);
  gemm(zq_bf, wb_erT, pr, nullptr, nullptr, 1024, 1024, 256, 1024, 256, 1024,
       1048576, 256, 0, 0, 4194304, 1048576, 4, 16, 0);
  softmax_kernel<<<dim3(16384), dim3(256), 0, stream>>>(qk, pr, A_bf);

  // a@v -> o_concat (bf16), then output projection
  gemm(A_bf, vT_bf, nullptr, oc_bf, nullptr, 1024, 256, 1024, 1024, 1024, 1024,
       4194304, 1048576, 1048576, 262144, 1048576, 256, 4, 16, 0);
  gemm(oc_bf, wb_o, o_out, nullptr, o_b, 4096, 1024, 1024, 1024, 1024, 1024,
       0, 0, 0, 0, 0, 0, 1, 1, EP_BIAS);

  // ln5 (x += o residual) -> FFN
  ln_kernel<<<dim3(4096), dim3(256), 0, stream>>>(x, o_out, 1, ln5_g, ln5_b, nullptr, h_bf);
  gemm(h_bf, wb_c2, nullptr, m_bf, nullptr, 4096, 2048, 1024, 1024, 1024, 2048,
       0, 0, 0, 0, 0, 0, 1, 1, EP_RELUSQ);
  gemm(m_bf, wb_c3, ff, nullptr, nullptr, 4096, 1024, 2048, 2048, 2048, 1024,
       0, 0, 0, 0, 0, 0, 1, 1, 0);

  // output: clip((src,x+ff) . out_w, 0, 6)/6, duplicate last frequency row
  final_kernel<<<dim3(32, 32, BB), dim3(256), 0, stream>>>(src, x, ff, out_w, out);
  lastrow_kernel<<<dim3(32), dim3(256), 0, stream>>>(out);
}
